// SlotAttention_89550068121605
// MI455X (gfx1250) — matrix operand order repaired and checked by reading, not compiled, not run on hardware
//
#include <hip/hip_runtime.h>

// Slot Attention for MI455X (gfx1250, wave32, WMMA).
// Heavy GEMMs (K/V projection, q·k^T dots, attn·V updates) use
// v_wmma_f32_16x16x32_f16 with f16 operands / fp32 accumulation.
// K kept row-major f16, V kept TRANSPOSED f16 (d-major) so every WMMA
// operand load in the hot attention loop is a contiguous 32-byte load.
// K/V total 64 MB -> L2-resident (192 MB) across the 3 iterations.

#define BATCH 64
#define NTOK  4096
#define DIM   64
#define NSLOT 7
#define HID   128
#define NITER 3
#define EPS_A  1e-8f
#define LNEPS  1e-5f
#define QSCALE 0.125f   // D^-0.5, D=64

typedef __attribute__((ext_vector_type(16))) _Float16 v16h;
typedef __attribute__((ext_vector_type(8)))  float    v8f;

__device__ __forceinline__ v8f wmma_f16(v16h a, v16h b, v8f c) {
  // D = A(16x32,f16) * B(32x16,f16) + C(16x16,f32)
  return __builtin_amdgcn_wmma_f32_16x16x32_f16(
      /*neg_a=*/false, a, /*neg_b=*/false, b,
      /*c_mod=*/(short)0, c, /*reuse_a=*/false, /*reuse_b=*/false);
}

__device__ __forceinline__ float sigmoidf_(float x) {
  return 1.0f / (1.0f + __expf(-x));
}

// ---------------------------------------------------------------------------
// slots = mu + sigma * noise
__global__ __launch_bounds__(256) void init_slots_kernel(
    const float* __restrict__ noise, const float* __restrict__ mu,
    const float* __restrict__ sigma, float* __restrict__ slots) {
  int idx = blockIdx.x * 256 + threadIdx.x;
  if (idx < BATCH * NSLOT * DIM) {
    int d = idx & (DIM - 1);
    slots[idx] = mu[d] + sigma[d] * noise[idx];
  }
}

// ---------------------------------------------------------------------------
// Fused LayerNorm(inputs) + K = x@Wk^T+bk (row-major f16)
//                          + V = x@Wv^T+bv (TRANSPOSED f16: vt[b][d][j]).
// 256 threads = 8 waves; each wave computes a 16-row x 64-col output tile
// for both K and V via WMMA (K-dim 64 = two chained 16x16x32 WMMAs).
__global__ __launch_bounds__(256) void lnkv_kernel(
    const float* __restrict__ x,
    const float* __restrict__ Wk, const float* __restrict__ bk,
    const float* __restrict__ Wv, const float* __restrict__ bv,
    const float* __restrict__ g_in, const float* __restrict__ be_in,
    _Float16* __restrict__ kf, _Float16* __restrict__ vt) {
  __shared__ _Float16 wk_l[DIM * DIM];
  __shared__ _Float16 wv_l[DIM * DIM];
  __shared__ _Float16 xln[8][16 * DIM];
  __shared__ float bk_l[DIM], bv_l[DIM], gi_l[DIM], be_l[DIM];

  const int t = threadIdx.x;
  for (int idx = t; idx < DIM * DIM; idx += 256) {
    wk_l[idx] = (_Float16)Wk[idx];
    wv_l[idx] = (_Float16)Wv[idx];
  }
  if (t < DIM) {
    bk_l[t] = bk[t]; bv_l[t] = bv[t];
    gi_l[t] = g_in[t]; be_l[t] = be_in[t];
  }
  __syncthreads();

  const int w = t >> 5, l = t & 31;
  const int lm = l & 15, g = l >> 4;

  // LayerNorm: two lanes per row (redundant stats, each writes half the row)
  {
    const long rg = (long)blockIdx.x * 128 + w * 16 + lm;
    const float* xr = x + rg * DIM;
    float s = 0.f;
    for (int d = 0; d < DIM; ++d) s += xr[d];
    float mean = s * (1.0f / DIM);
    float v2 = 0.f;
    for (int d = 0; d < DIM; ++d) { float df = xr[d] - mean; v2 += df * df; }
    float rstd = rsqrtf(v2 * (1.0f / DIM) + LNEPS);
    for (int e = 0; e < 32; ++e) {
      int d = g * 32 + e;
      xln[w][lm * DIM + d] = (_Float16)((xr[d] - mean) * rstd * gi_l[d] + be_l[d]);
    }
  }
  __syncthreads();

  // A fragments (16x32 f16, contiguous k order shared with B): lane m=l&15;
  // half hi -> K = g*16+hi
  v16h a0, a1;
#pragma unroll
  for (int hi = 0; hi < 16; ++hi) {
    int kl = g * 16 + hi;
    a0[hi] = xln[w][lm * DIM + kl];
    a1[hi] = xln[w][lm * DIM + 32 + kl];
  }

  const long rowbase = (long)blockIdx.x * 128 + w * 16;  // 128 rows stay in one batch
  const long bb = rowbase >> 12;                          // batch index
  const long jb0 = rowbase & (NTOK - 1);                  // token index in batch
#pragma unroll
  for (int mat = 0; mat < 2; ++mat) {
    const _Float16* Wl = mat ? wv_l : wk_l;
    const float* bl = mat ? bv_l : bk_l;
#pragma unroll
    for (int ct = 0; ct < 4; ++ct) {
      // B fragment (32x16): lane n=l&15; half i -> K = g*16+i
      const int c = ct * 16 + lm;
      v16h b0, b1;
#pragma unroll
      for (int i = 0; i < 16; ++i) {
        b0[i] = Wl[c * DIM + g * 16 + i];
        b1[i] = Wl[c * DIM + 32 + g * 16 + i];
      }
      float bias = bl[c];
      v8f acc;
#pragma unroll
      for (int r = 0; r < 8; ++r) acc[r] = bias;
      acc = wmma_f16(a0, b0, acc);
      acc = wmma_f16(a1, b1, acc);
#pragma unroll
      for (int r = 0; r < 8; ++r) {
        int m = g * 8 + r;           // C layout: lanes0-15 -> M=r, lanes16-31 -> M=8+r
        if (mat == 0) {
          kf[(rowbase + m) * DIM + c] = (_Float16)acc[r];
        } else {
          // transposed: vt[(b*64 + d)*NTOK + j]
          vt[(bb * DIM + c) * NTOK + jb0 + m] = (_Float16)acc[r];
        }
      }
    }
  }
}

// ---------------------------------------------------------------------------
// q = LN(slots)@Wq^T + bq, written zero-padded to 16 rows per batch (f16).
__global__ __launch_bounds__(64) void qproj_kernel(
    const float* __restrict__ slots, const float* __restrict__ Wq,
    const float* __restrict__ bq, const float* __restrict__ g_sl,
    const float* __restrict__ be_sl, _Float16* __restrict__ qpad) {
  const int b = blockIdx.x >> 4, i = blockIdx.x & 15, t = threadIdx.x;
  _Float16* qrow = qpad + ((long)b * 16 + i) * DIM;
  if (i >= NSLOT) { qrow[t] = (_Float16)0.f; return; }
  __shared__ float red[DIM], sx[DIM], stats[2];
  const float* sr = slots + ((long)b * NSLOT + i) * DIM;
  float v = sr[t];
  red[t] = v; __syncthreads();
  for (int s = 32; s > 0; s >>= 1) { if (t < s) red[t] += red[t + s]; __syncthreads(); }
  if (t == 0) stats[0] = red[0] * (1.0f / DIM);
  __syncthreads();
  float df = v - stats[0];
  red[t] = df * df; __syncthreads();
  for (int s = 32; s > 0; s >>= 1) { if (t < s) red[t] += red[t + s]; __syncthreads(); }
  if (t == 0) stats[1] = rsqrtf(red[0] * (1.0f / DIM) + LNEPS);
  __syncthreads();
  sx[t] = df * stats[1] * g_sl[t] + be_sl[t];
  __syncthreads();
  float acc = bq[t];
  for (int d = 0; d < DIM; ++d) acc += sx[d] * Wq[t * DIM + d];
  qrow[t] = (_Float16)acc;
}

// ---------------------------------------------------------------------------
// Per-batch attention: dots = q@k^T*scale, softmax over slot axis (NS=7,
// lane-local: slots 0..6 sit in C VGPRs 0..6 of lanes 0-15), +EPS,
// updates[i] = (sum_j a[i][j]*v[j]) / (sum_j a[i][j])  via WMMA.
// All WMMA operand loads are contiguous 32B per lane (K row-major, V d-major).
__global__ __launch_bounds__(256) void attn_kernel(
    const _Float16* __restrict__ qpad, const _Float16* __restrict__ kf,
    const _Float16* __restrict__ vt, float* __restrict__ updates) {
  const int bidx = blockIdx.x;
  const int t = threadIdx.x, w = t >> 5, l = t & 31, lm = l & 15, g = l >> 4;
  __shared__ _Float16 q_lds[16 * DIM];
  __shared__ _Float16 att[8][16 * 32];     // per-wave attn tile, A-layout source
  __shared__ float upd[NSLOT * DIM];
  __shared__ float asum_s[NSLOT];

  for (int idx = t; idx < 16 * DIM; idx += 256)
    q_lds[idx] = qpad[(long)bidx * 16 * DIM + idx];
  for (int idx = t; idx < 8 * 16 * 32; idx += 256)
    (&att[0][0])[idx] = (_Float16)0.f;     // rows 7..15 stay zero forever
  for (int idx = t; idx < NSLOT * DIM; idx += 256) upd[idx] = 0.f;
  if (t < NSLOT) asum_s[t] = 0.f;
  __syncthreads();

  v16h aq0, aq1;
#pragma unroll
  for (int hi = 0; hi < 16; ++hi) {
    int kl = g * 16 + hi;
    aq0[hi] = q_lds[lm * DIM + kl];
    aq1[hi] = q_lds[lm * DIM + 32 + kl];
  }

  v8f cu[4];
#pragma unroll
  for (int ct = 0; ct < 4; ++ct)
#pragma unroll
    for (int r = 0; r < 8; ++r) cu[ct][r] = 0.f;

  float asum_r[NSLOT];
#pragma unroll
  for (int i = 0; i < NSLOT; ++i) asum_r[i] = 0.f;

  const long kvbase = (long)bidx * NTOK;
  const _Float16* vtb = vt + (long)bidx * DIM * NTOK;
  for (int ci = 0; ci < 16; ++ci) {
    const int j0 = (ci * 8 + w) * 32;        // 16 chunks x 8 waves x 32 j = 4096

    if (ci < 15) {                           // pull next chunk's lines toward WGP
      const int j0n = j0 + 256;
      __builtin_prefetch(kf + (kvbase + j0n + l) * DIM, 0, 3);
      __builtin_prefetch(vtb + (long)l * NTOK + j0n, 0, 3);
      __builtin_prefetch(vtb + (long)(l + 32) * NTOK + j0n, 0, 3);
    }

#pragma unroll
    for (int jt = 0; jt < 2; ++jt) {
      const int jb = j0 + jt * 16;
      const _Float16* kr = kf + (kvbase + jb + lm) * DIM;
      v16h b0, b1;                           // B[kk][n=j] = k[jb+n][kk], contiguous
#pragma unroll
      for (int i = 0; i < 16; ++i) { b0[i] = kr[g * 16 + i]; b1[i] = kr[32 + g * 16 + i]; }
      v8f c;
#pragma unroll
      for (int r = 0; r < 8; ++r) c[r] = 0.f;
      c = wmma_f16(aq0, b0, c);
      c = wmma_f16(aq1, b1, c);
      if (l < 16) {                          // lane l holds dots[m=0..7][j=jb+l]
        float vv[NSLOT];
#pragma unroll
        for (int i = 0; i < NSLOT; ++i) vv[i] = c[i] * QSCALE;
        float mx = vv[0];
#pragma unroll
        for (int i = 1; i < NSLOT; ++i) mx = fmaxf(mx, vv[i]);
        float e[NSLOT]; float ssum = 0.f;
#pragma unroll
        for (int i = 0; i < NSLOT; ++i) { e[i] = __expf(vv[i] - mx); ssum += e[i]; }
        float inv = 1.0f / ssum;
#pragma unroll
        for (int i = 0; i < NSLOT; ++i) {
          float a = e[i] * inv + EPS_A;
          asum_r[i] += a;
          att[w][i * 32 + jt * 16 + lm] = (_Float16)a;
        }
      }
    }
    __syncthreads();   // make att tile visible (and keep compiler honest)

    v16h av;           // attn as A-matrix (16 slots x 32 j)
#pragma unroll
    for (int hi = 0; hi < 16; ++hi) {
      int kl = g * 16 + hi;
      av[hi] = att[w][lm * 32 + kl];
    }
#pragma unroll
    for (int ct = 0; ct < 4; ++ct) {
      // B[kk=j_local][n=d] = vt row d, contiguous 16 halves at j0+g*16
      const _Float16* vr = vtb + (long)(ct * 16 + lm) * NTOK + j0 + g * 16;
      v16h bvv;
#pragma unroll
      for (int i = 0; i < 16; ++i) bvv[i] = vr[i];
      cu[ct] = wmma_f16(av, bvv, cu[ct]);
    }
  }

  // cross-wave reduce: lanes 0-15 hold slot rows 0..7 (7 valid)
  if (g == 0) {
#pragma unroll
    for (int ct = 0; ct < 4; ++ct)
#pragma unroll
      for (int r = 0; r < NSLOT; ++r)
        atomicAdd(&upd[r * DIM + ct * 16 + lm], cu[ct][r]);
#pragma unroll
    for (int i = 0; i < NSLOT; ++i) atomicAdd(&asum_s[i], asum_r[i]);
  }
  __syncthreads();
  for (int idx = t; idx < NSLOT * DIM; idx += 256)
    updates[(long)bidx * NSLOT * DIM + idx] = upd[idx] / asum_s[idx >> 6];
}

// ---------------------------------------------------------------------------
// GRU cell (torch semantics), one thread per (row, d). 448x64 — tiny.
__global__ __launch_bounds__(256) void gru_kernel(
    const float* __restrict__ updates, const float* __restrict__ slots_in,
    const float* __restrict__ W_ih, const float* __restrict__ W_hh,
    const float* __restrict__ b_ih, const float* __restrict__ b_hh,
    float* __restrict__ slots_out) {
  int idx = blockIdx.x * 256 + threadIdx.x;
  if (idx >= BATCH * NSLOT * DIM) return;
  int r = idx >> 6, d = idx & (DIM - 1);
  const float* xr = updates + (long)r * DIM;
  const float* hr = slots_in + (long)r * DIM;
  float gi[3], gh[3];
#pragma unroll
  for (int p = 0; p < 3; ++p) {
    const float* wi = W_ih + (long)(p * DIM + d) * DIM;
    const float* wh = W_hh + (long)(p * DIM + d) * DIM;
    float ai = b_ih[p * DIM + d], ah = b_hh[p * DIM + d];
    for (int k = 0; k < DIM; ++k) { ai += xr[k] * wi[k]; ah += hr[k] * wh[k]; }
    gi[p] = ai; gh[p] = ah;
  }
  float rg = sigmoidf_(gi[0] + gh[0]);
  float z  = sigmoidf_(gi[1] + gh[1]);
  float nn = tanhf(gi[2] + rg * gh[2]);
  slots_out[idx] = (1.0f - z) * nn + z * hr[d];
}

// ---------------------------------------------------------------------------
// slots += relu(LN(slots)@W1^T + b1)@W2^T + b2 ; one block per slot row.
__global__ __launch_bounds__(128) void ff_kernel(
    const float* __restrict__ slots_in, const float* __restrict__ W1,
    const float* __restrict__ b1, const float* __restrict__ W2,
    const float* __restrict__ b2, const float* __restrict__ g_ff,
    const float* __restrict__ be_ff, float* __restrict__ slots_out) {
  const int row = blockIdx.x, t = threadIdx.x;
  __shared__ float srow[DIM], sx[DIM], hid[HID], red[DIM], stats[2];
  const float* sr = slots_in + (long)row * DIM;
  if (t < DIM) { srow[t] = sr[t]; red[t] = srow[t]; }
  __syncthreads();
  for (int s = 32; s > 0; s >>= 1) { if (t < s) red[t] += red[t + s]; __syncthreads(); }
  if (t == 0) stats[0] = red[0] * (1.0f / DIM);
  __syncthreads();
  if (t < DIM) { float df = srow[t] - stats[0]; red[t] = df * df; }
  __syncthreads();
  for (int s = 32; s > 0; s >>= 1) { if (t < s) red[t] += red[t + s]; __syncthreads(); }
  if (t == 0) stats[1] = rsqrtf(red[0] * (1.0f / DIM) + LNEPS);
  __syncthreads();
  if (t < DIM) sx[t] = (srow[t] - stats[0]) * stats[1] * g_ff[t] + be_ff[t];
  __syncthreads();
  {
    float acc = b1[t];
    const float* w1 = W1 + (long)t * DIM;
    for (int d = 0; d < DIM; ++d) acc += sx[d] * w1[d];
    hid[t] = fmaxf(acc, 0.f);
  }
  __syncthreads();
  if (t < DIM) {
    float acc = b2[t];
    const float* w2 = W2 + (long)t * HID;
    for (int h2 = 0; h2 < HID; ++h2) acc += hid[h2] * w2[h2];
    slots_out[(long)row * DIM + t] = srow[t] + acc;
  }
}

// ---------------------------------------------------------------------------
extern "C" void kernel_launch(void* const* d_in, const int* in_sizes, int n_in,
                              void* d_out, int out_size, void* d_ws, size_t ws_size,
                              hipStream_t stream) {
  (void)in_sizes; (void)n_in; (void)out_size; (void)ws_size;
  const float* inputs = (const float*)d_in[0];
  // d_in[1] = positional_embeddings (unused by reference)
  const float* noise  = (const float*)d_in[2];
  const float* mu     = (const float*)d_in[3];
  const float* sigma  = (const float*)d_in[4];
  const float* Wq     = (const float*)d_in[5];
  const float* bq     = (const float*)d_in[6];
  const float* Wk     = (const float*)d_in[7];
  const float* bk     = (const float*)d_in[8];
  const float* Wv     = (const float*)d_in[9];
  const float* bv     = (const float*)d_in[10];
  const float* W_ih   = (const float*)d_in[11];
  const float* W_hh   = (const float*)d_in[12];
  const float* b_ih   = (const float*)d_in[13];
  const float* b_hh   = (const float*)d_in[14];
  const float* W1     = (const float*)d_in[15];
  const float* b1     = (const float*)d_in[16];
  const float* W2     = (const float*)d_in[17];
  const float* b2     = (const float*)d_in[18];
  const float* g_in   = (const float*)d_in[19];
  const float* be_in  = (const float*)d_in[20];
  const float* g_sl   = (const float*)d_in[21];
  const float* be_sl  = (const float*)d_in[22];
  const float* g_ff   = (const float*)d_in[23];
  const float* be_ff  = (const float*)d_in[24];

  // Workspace layout (bytes), 256-aligned offsets; total ~67.6 MB.
  char* ws = (char*)d_ws;
  float*    slotsA  = (float*)(ws + 0);                   // 114688
  float*    slotsB  = (float*)(ws + 114688);              // 114688
  float*    updates = (float*)(ws + 229376);              // 114688
  _Float16* qpad    = (_Float16*)(ws + 344064);           // 131072
  _Float16* kf      = (_Float16*)(ws + 475136);           // 33554432
  _Float16* vt      = (_Float16*)(ws + 475136 + 33554432);// 33554432 (transposed V)

  init_slots_kernel<<<(BATCH * NSLOT * DIM + 255) / 256, 256, 0, stream>>>(
      noise, mu, sigma, slotsA);
  lnkv_kernel<<<(BATCH * NTOK) / 128, 256, 0, stream>>>(
      inputs, Wk, bk, Wv, bv, g_in, be_in, kf, vt);

  for (int it = 0; it < NITER; ++it) {
    qproj_kernel<<<BATCH * 16, 64, 0, stream>>>(slotsA, Wq, bq, g_sl, be_sl, qpad);
    attn_kernel<<<BATCH, 256, 0, stream>>>(qpad, kf, vt, updates);
    gru_kernel<<<(BATCH * NSLOT * DIM + 255) / 256, 256, 0, stream>>>(
        updates, slotsA, W_ih, W_hh, b_ih, b_hh, slotsB);
    ff_kernel<<<BATCH * NSLOT, 128, 0, stream>>>(
        slotsB, W1, b1, W2, b2, g_ff, be_ff, slotsA);
  }

  hipMemcpyAsync(d_out, slotsA, (size_t)BATCH * NSLOT * DIM * sizeof(float),
                 hipMemcpyDeviceToDevice, stream);
}
